// EdgeUpdateNetwork_13864154432274
// MI455X (gfx1250) — matrix operand order repaired and checked by reading, not compiled, not run on hardware
//
#include <hip/hip_runtime.h>
#include <hip/hip_bf16.h>

typedef __bf16 bf16t;
typedef __attribute__((ext_vector_type(16))) __bf16 v16bf;
typedef __attribute__((ext_vector_type(8)))  __bf16 v8bf;
typedef __attribute__((ext_vector_type(8)))  float  v8f;

constexpr int E      = 160000;
constexpr int NNODES = 10000;

// ---------------- device helpers ----------------

__device__ __forceinline__ bf16t f2bf(float f) { return (bf16t)f; }   // native v_cvt, RTNE
__device__ __forceinline__ unsigned short bfbits(float f) {
    return __builtin_bit_cast(unsigned short, (bf16t)f);
}
// two floats -> packed bf16x2 (one u32 LDS store, pairable to v_cvt_pk_bf16_f32)
__device__ __forceinline__ unsigned pack2(float a, float b) {
    return (unsigned)bfbits(a) | ((unsigned)bfbits(b) << 16);
}
// 1-ulp hardware ops (v_rcp_f32 / v_sqrt_f32) instead of IEEE div/sqrt expansions
__device__ __forceinline__ float frcp(float x)  { return __builtin_amdgcn_rcpf(x); }
__device__ __forceinline__ float fsqrt(float x) { return __builtin_amdgcn_sqrtf(x); }
__device__ __forceinline__ float silu(float x) { return x * frcp(1.f + __expf(-x)); }
__device__ __forceinline__ float sigm(float x) { return frcp(1.f + __expf(-x)); }

union V16U { v16bf v; v8bf h[2]; };

// A tile 16x32 bf16 from LDS. Layout: lanes0-15 row M=lane hold K {k0..k0+15},
// lanes16-31 row M=lane-16 hold K {k0+16..k0+31}.
__device__ __forceinline__ v16bf load_a(const bf16t* As, int ldA, int k0, int lane) {
    int r  = lane & 15;
    int c0 = k0 + ((lane >> 4) << 4);
    const bf16t* row = As + r * ldA;
    V16U t;
    t.h[0] = *(const v8bf*)(row + c0);
    t.h[1] = *(const v8bf*)(row + c0 + 8);
    return t.v;
}
// B tile 32x16 bf16 from global Wt[n][kpad] (transposed, K-contiguous 32B load per lane).
__device__ __forceinline__ v16bf load_b(const bf16t* Wt, int kpad, int n0, int k0, int lane) {
    int col = lane & 15;
    int kb  = k0 + ((lane >> 4) << 4);
    return *(const v16bf*)(Wt + (size_t)(n0 + col) * kpad + kb);
}

// A-fragment set for a whole GEMM stage: loaded from LDS once, reused across all N-tiles.
template<int K>
struct AF { v16bf f[K / 32]; };

template<int K>
__device__ __forceinline__ AF<K> load_af(const bf16t* As, int ldA, int lane) {
    AF<K> r;
#pragma unroll
    for (int k = 0; k < K / 32; ++k) r.f[k] = load_a(As, ldA, 32 * k, lane);
    return r;
}
template<int K>
__device__ __forceinline__ v8f wmma_af(const AF<K>& A, const bf16t* Wt, int kpad,
                                       int n0, int lane, v8f acc) {
#pragma unroll
    for (int k = 0; k < K / 32; ++k)
        acc = __builtin_amdgcn_wmma_f32_16x16x32_bf16(false, A.f[k], false,
                load_b(Wt, kpad, n0, 32 * k, lane), (short)0, acc, false, false);
    return acc;
}

// C layout: acc[j] = C[(lane>=16 ? 8 : 0) + j][lane&15]
__device__ __forceinline__ void store_bf16(v8f acc, bf16t* C, int ldc, int n0,
                                           const float* bias, int lane) {
    int col = n0 + (lane & 15);
    int rb  = (lane >> 4) * 8;
    float b = bias ? bias[col] : 0.f;
#pragma unroll
    for (int j = 0; j < 8; ++j)
        C[(rb + j) * ldc + col] = f2bf(acc[j] + b);
}

// ---------------- weight prep: fp32 [K][N] -> bf16 transposed padded [npad][kpad] ----------------

__global__ void conv_w(const float* __restrict__ src, bf16t* __restrict__ dst,
                       int K, int N, int kpad, int npad) {
    int idx = blockIdx.x * blockDim.x + threadIdx.x;
    int total = kpad * npad;
    if (idx >= total) return;
    int n = idx / kpad, k = idx % kpad;
    float v = (k < K && n < N) ? src[(size_t)k * N + n] : 0.f;
    dst[idx] = f2bf(v);
}

// ---------------- K2: fused = silu([e_s|rbf] @ W1 + b1) @ W2 + b2  (stored bf16) ----------------

__global__ void __launch_bounds__(128) fused_mlp_kernel(
    const float* __restrict__ eIn, const float* __restrict__ rbf,
    const bf16t* __restrict__ W1t, const float* __restrict__ b1,
    const bf16t* __restrict__ W2t, const float* __restrict__ b2,
    bf16t* __restrict__ hbS)
{
    __shared__ __align__(128) bf16t AxS[4][16 * 160];
    __shared__ __align__(128) bf16t AmS[4][16 * 256];
    int lane = threadIdx.x & 31, w = threadIdx.x >> 5;
    bf16t* Ax = AxS[w];
    bf16t* Am = AmS[w];
    int e0 = (blockIdx.x * 4 + w) * 16;

#pragma unroll 4
    for (int r = 0; r < 16; ++r) {
        int e = e0 + r;
        const float* xs = eIn + (size_t)e * 320;
        const float* xr = rbf + (size_t)e * 32;
        unsigned* dst = (unsigned*)(Ax + r * 160);       // 80 u32 slots
        dst[lane]      = pack2(xs[2 * lane],      xs[2 * lane + 1]);
        dst[32 + lane] = pack2(xs[2 * lane + 64], xs[2 * lane + 65]);
        if (lane < 16) dst[64 + lane] = pack2(xr[2 * lane], xr[2 * lane + 1]);
    }
    __syncthreads();
    {
        AF<160> A = load_af<160>(Ax, 160, lane);
#pragma unroll
        for (int nt = 0; nt < 16; ++nt) {
            v8f z = {};
            v8f acc = wmma_af<160>(A, W1t, 160, nt * 16, lane, z);
            int col = nt * 16 + (lane & 15), rb = (lane >> 4) * 8;
            float b = b1[col];
#pragma unroll
            for (int j = 0; j < 8; ++j)
                Am[(rb + j) * 256 + col] = f2bf(silu(acc[j] + b));
        }
    }
    __syncthreads();
    {
        AF<256> A = load_af<256>(Am, 256, lane);
#pragma unroll
        for (int nt = 0; nt < 8; ++nt) {
            v8f z = {};
            v8f acc = wmma_af<256>(A, W2t, 256, nt * 16, lane, z);
            int col = nt * 16 + (lane & 15), rb = (lane >> 4) * 8;
            float b = b2[col];
#pragma unroll
            for (int j = 0; j < 8; ++j)
                hbS[(size_t)(e0 + rb + j) * 128 + col] = f2bf(acc[j] + b);
        }
    }
}

// ---------------- K3: fused bond_ffn + scatter-add ----------------

struct FfnW {
    const bf16t *bondWs, *bondWv, *nodeWs, *nodeWv;
    const bf16t *fus1Ws, *fus1Wv, *fus2Ws, *fus2Wv, *gate1Ws;
    const float *bondB, *nodeB, *fus1B, *fus2B, *gate1B, *gate2W, *gate2B;
};

__global__ void __launch_bounds__(32) bond_ffn_kernel(
    const bf16t* __restrict__ hbS, const float* __restrict__ eIn,
    const float* __restrict__ h,   const float* __restrict__ eattr,
    const int* __restrict__ gidx,  const int* __restrict__ aidx,
    float* __restrict__ agg, FfnW W)
{
    // LDS map (48KB): Ags 16x288 | region{ Agv 3x16x160 + Acv 3x16x96  /  Fs 16x128 f32 + Fv 3x16x64 f32 }
    //                 | Acs 16x160 (later Gd 16x16 f32 + gate) | Afs 16x128 | Afv 3x16x64
    __shared__ __align__(128) char smem[49152];
    bf16t* Ags  = (bf16t*)(smem);
    bf16t* Agv  = (bf16t*)(smem + 9216);
    bf16t* Acv  = (bf16t*)(smem + 9216 + 15360);
    bf16t* Acs  = (bf16t*)(smem + 33792);
    bf16t* Afs  = (bf16t*)(smem + 38912);
    bf16t* Afv  = (bf16t*)(smem + 43008);
    float* Fs   = (float*)(smem + 9216);
    float* Fv   = (float*)(smem + 9216 + 8192);
    float* Gd   = (float*)(smem + 33792);
    float* gate = (float*)(smem + 33792 + 1024);

    const int lane = threadIdx.x;
    const int e0   = blockIdx.x * 16;

    // ---- staging, one pass per edge row; gather index hoisted per row; packed u32 LDS stores ----
    for (int r = 0; r < 16; ++r) {
        int e   = e0 + r;
        int nid = gidx[e];
        const float* hn = h + (size_t)nid * 320;

        // gs row = [hb_s(128) | h_s(128) | ss | 0pad]  (288 el = 144 u32)
        {
            const unsigned* hs = (const unsigned*)(hbS + (size_t)e * 128);   // already bf16
            unsigned* dg = (unsigned*)(Ags + r * 288);
            dg[lane]      = hs[lane];
            dg[32 + lane] = hs[32 + lane];
            dg[64 + lane] = pack2(hn[2 * lane],      hn[2 * lane + 1]);
            dg[96 + lane] = pack2(hn[2 * lane + 64], hn[2 * lane + 65]);
            if (lane == 0)      dg[128]        = (unsigned)bfbits(eattr[(size_t)e * 4]); // col256=ss, col257=0
            else if (lane < 16) dg[128 + lane] = 0u;                                     // cols 258..287
        }
        // gv rows, comps i=0..2 = [hb_v_i(64) | h_v_i(64) | sv_i | 0pad] (160 el = 80 u32)
#pragma unroll
        for (int i = 0; i < 3; ++i) {
            const float* ev = eIn + (size_t)e * 320 + 128 + i;
            const float* hv = hn + 128 + i;
            unsigned* dv = (unsigned*)(Agv + i * 2560 + r * 160);
            dv[lane]      = pack2(ev[6 * lane], ev[6 * lane + 3]);
            dv[32 + lane] = pack2(hv[6 * lane], hv[6 * lane + 3]);
            if (lane == 0)      dv[64]        = (unsigned)bfbits(eattr[(size_t)e * 4 + 1 + i]);
            else if (lane < 16) dv[64 + lane] = 0u;                                      // cols 130..159
        }
        // cs pad row: cols 128..159 (u32 slots 64..79): slot64 = (ss, 0)
        {
            unsigned* dc = (unsigned*)(Acs + r * 160);
            if (lane == 0)      dc[64]        = (unsigned)bfbits(eattr[(size_t)e * 4]);
            else if (lane < 16) dc[64 + lane] = 0u;
        }
        // cv pad rows: cols 64..95 (u32 slots 32..47): slot32 = (sv_i, 0)
#pragma unroll
        for (int i = 0; i < 3; ++i) {
            unsigned* dcv = (unsigned*)(Acv + i * 1536 + r * 96);
            if (lane == 0)      dcv[32]        = (unsigned)bfbits(eattr[(size_t)e * 4 + 1 + i]);
            else if (lane < 16) dcv[32 + lane] = 0u;
        }
    }
    __syncthreads();

    // bond/node scalar elins -> cs   (A-frags hoisted, reused across N-tiles)
    {
        AF<128> Ab = load_af<128>(Ags,       288, lane);
        AF<128> An = load_af<128>(Ags + 128, 288, lane);
#pragma unroll
        for (int nt = 0; nt < 4; ++nt) {
            v8f z = {};
            store_bf16(wmma_af<128>(Ab, W.bondWs, 128, nt * 16, lane, z), Acs,      160, nt * 16, W.bondB, lane);
            store_bf16(wmma_af<128>(An, W.nodeWs, 128, nt * 16, lane, z), Acs + 64, 160, nt * 16, W.nodeB, lane);
        }
    }
    // bond/node vector elins -> cv
#pragma unroll
    for (int i = 0; i < 3; ++i) {
        AF<64> Ab = load_af<64>(Agv + i * 2560,      160, lane);
        AF<64> An = load_af<64>(Agv + i * 2560 + 64, 160, lane);
        bf16t* Ci = Acv + i * 1536;
#pragma unroll
        for (int nt = 0; nt < 2; ++nt) {
            v8f z = {};
            store_bf16(wmma_af<64>(Ab, W.bondWv, 64, nt * 16, lane, z), Ci,      96, nt * 16, nullptr, lane);
            store_bf16(wmma_af<64>(An, W.nodeWv, 64, nt * 16, lane, z), Ci + 32, 96, nt * 16, nullptr, lane);
        }
    }
    __syncthreads();

    // fus1 + norm_act
    {
        AF<160> A = load_af<160>(Acs, 160, lane);
#pragma unroll
        for (int nt = 0; nt < 8; ++nt) {
            v8f z = {};
            v8f acc = wmma_af<160>(A, W.fus1Ws, 160, nt * 16, lane, z);
            int col = nt * 16 + (lane & 15), rb = (lane >> 4) * 8;
            float b = W.fus1B[col];
#pragma unroll
            for (int j = 0; j < 8; ++j)
                Afs[(rb + j) * 128 + col] = f2bf(silu(acc[j] + b));
        }
    }
    {
        AF<96> A0 = load_af<96>(Acv,        96, lane);
        AF<96> A1 = load_af<96>(Acv + 1536, 96, lane);
        AF<96> A2 = load_af<96>(Acv + 3072, 96, lane);
#pragma unroll
        for (int nt = 0; nt < 4; ++nt) {
            v8f z = {};
            v8f a0 = wmma_af<96>(A0, W.fus1Wv, 96, nt * 16, lane, z);
            v8f a1 = wmma_af<96>(A1, W.fus1Wv, 96, nt * 16, lane, z);
            v8f a2 = wmma_af<96>(A2, W.fus1Wv, 96, nt * 16, lane, z);
            int col = nt * 16 + (lane & 15), rb = (lane >> 4) * 8;
#pragma unroll
            for (int j = 0; j < 8; ++j) {
                float n  = fsqrt(a0[j] * a0[j] + a1[j] * a1[j] + a2[j] * a2[j]);
                float sc = silu(n) * frcp(n + 1e-8f);
                Afv[       (rb + j) * 64 + col] = f2bf(a0[j] * sc);
                Afv[1024 + (rb + j) * 64 + col] = f2bf(a1[j] * sc);
                Afv[2048 + (rb + j) * 64 + col] = f2bf(a2[j] * sc);
            }
        }
    }
    __syncthreads();

    // fus2 -> Fs/Fv (f32, overlays dead Agv/Acv)
    {
        AF<128> A = load_af<128>(Afs, 128, lane);
#pragma unroll
        for (int nt = 0; nt < 8; ++nt) {
            v8f z = {};
            v8f acc = wmma_af<128>(A, W.fus2Ws, 128, nt * 16, lane, z);
            int col = nt * 16 + (lane & 15), rb = (lane >> 4) * 8;
            float b = W.fus2B[col];
#pragma unroll
            for (int j = 0; j < 8; ++j) Fs[(rb + j) * 128 + col] = acc[j] + b;
        }
    }
#pragma unroll
    for (int i = 0; i < 3; ++i) {
        AF<64> A = load_af<64>(Afv + i * 1024, 64, lane);
#pragma unroll
        for (int nt = 0; nt < 4; ++nt) {
            v8f z = {};
            v8f acc = wmma_af<64>(A, W.fus2Wv, 64, nt * 16, lane, z);
            int col = nt * 16 + (lane & 15), rb = (lane >> 4) * 8;
#pragma unroll
            for (int j = 0; j < 8; ++j) Fv[i * 1024 + (rb + j) * 64 + col] = acc[j];
        }
    }

    // gate: sigmoid( silu(gs @ gate1.Ws + b) . gate2W + gate2b )   (g1v branch is discarded -> skipped)
    {
        AF<288> A = load_af<288>(Ags, 288, lane);
        float p[8] = {0, 0, 0, 0, 0, 0, 0, 0};
        for (int nt = 0; nt < 17; ++nt) {
            v8f z = {};
            v8f acc = wmma_af<288>(A, W.gate1Ws, 288, nt * 16, lane, z);
            int col  = nt * 16 + (lane & 15);
            float b  = (col < 257) ? W.gate1B[col] : 0.f;
            float w2 = (col < 257) ? W.gate2W[col] : 0.f;
#pragma unroll
            for (int j = 0; j < 8; ++j) p[j] += silu(acc[j] + b) * w2;
        }
        int col = lane & 15, rb = (lane >> 4) * 8;
#pragma unroll
        for (int j = 0; j < 8; ++j) Gd[(rb + j) * 16 + col] = p[j];
    }
    __syncthreads();
    if (lane < 16) {
        float s = 0.f;
        for (int c = 0; c < 16; ++c) s += Gd[lane * 16 + c];
        gate[lane] = sigm(s + W.gate2B[0]);
    }
    __syncthreads();

    // scatter-add message into agg[aidx]; row scalars hoisted
    for (int r = 0; r < 16; ++r) {
        int e = e0 + r;
        float g = gate[r];
        float* ag = agg + (size_t)aidx[e] * 320;
        for (int c = lane; c < 128; c += 32)
            atomicAdd(ag + c, Fs[r * 128 + c] * g);
        for (int c = lane; c < 192; c += 32)
            atomicAdd(ag + 128 + c, Fv[(c % 3) * 1024 + r * 64 + (c / 3)] * g);
    }
}

// ---------------- K4: out = agg_l[left] + agg_r[right] + node_l + node_r + self ----------------

__global__ void __launch_bounds__(64) finalize_kernel(
    const float* __restrict__ h, const bf16t* __restrict__ hbS, const float* __restrict__ eIn,
    const int* __restrict__ left, const int* __restrict__ right,
    const float* __restrict__ aggL, const float* __restrict__ aggR,
    const bf16t* WlS, const float* blS, const bf16t* WlV,
    const bf16t* WrS, const float* brS, const bf16t* WrV,
    const bf16t* WbS, const float* bbS, const bf16t* WbV,
    float* __restrict__ out)
{
    __shared__ __align__(128) bf16t S4[2][3][16 * 128];
    __shared__ __align__(128) bf16t V4[2][3][3 * 16 * 64];
    int w = threadIdx.x >> 5, lane = threadIdx.x & 31;
    int e0 = (blockIdx.x * 2 + w) * 16;
    bf16t *Asl = S4[w][0], *Asr = S4[w][1], *Asb = S4[w][2];
    bf16t *Avl = V4[w][0], *Avr = V4[w][1], *Avb = V4[w][2];

    for (int r = 0; r < 16; ++r) {
        int e = e0 + r;
        const float* hl = h + (size_t)left[e]  * 320;
        const float* hr = h + (size_t)right[e] * 320;
        unsigned* dl = (unsigned*)(Asl + r * 128);
        unsigned* dr = (unsigned*)(Asr + r * 128);
        unsigned* db = (unsigned*)(Asb + r * 128);
        const unsigned* hb = (const unsigned*)(hbS + (size_t)e * 128);
        dl[lane]      = pack2(hl[2 * lane],      hl[2 * lane + 1]);
        dl[32 + lane] = pack2(hl[2 * lane + 64], hl[2 * lane + 65]);
        dr[lane]      = pack2(hr[2 * lane],      hr[2 * lane + 1]);
        dr[32 + lane] = pack2(hr[2 * lane + 64], hr[2 * lane + 65]);
        db[lane]      = hb[lane];
        db[32 + lane] = hb[32 + lane];
#pragma unroll
        for (int i = 0; i < 3; ++i) {
            const float* eb = eIn + (size_t)e * 320 + 128 + i;
            ((unsigned*)(Avl + i * 1024 + r * 64))[lane] = pack2(hl[128 + 6 * lane + i], hl[131 + 6 * lane + i]);
            ((unsigned*)(Avr + i * 1024 + r * 64))[lane] = pack2(hr[128 + 6 * lane + i], hr[131 + 6 * lane + i]);
            ((unsigned*)(Avb + i * 1024 + r * 64))[lane] = pack2(eb[6 * lane],           eb[6 * lane + 3]);
        }
    }
    __syncthreads();

    // per-lane row pointers hoisted out of the tile loops
    int rb = (lane >> 4) * 8;
    const float* aLp[8];
    const float* aRp[8];
    float*       op[8];
#pragma unroll
    for (int j = 0; j < 8; ++j) {
        int e = e0 + rb + j;
        aLp[j] = aggL + (size_t)left[e]  * 320;
        aRp[j] = aggR + (size_t)right[e] * 320;
        op[j]  = out  + (size_t)e * 320;
    }

    // scalar part: accumulate the three sources into persistent per-N-tile accumulators
    {
        v8f acc[8];
#pragma unroll
        for (int nt = 0; nt < 8; ++nt) acc[nt] = v8f{};
        {
            AF<128> A = load_af<128>(Asl, 128, lane);
#pragma unroll
            for (int nt = 0; nt < 8; ++nt) acc[nt] = wmma_af<128>(A, WlS, 128, nt * 16, lane, acc[nt]);
        }
        {
            AF<128> A = load_af<128>(Asr, 128, lane);
#pragma unroll
            for (int nt = 0; nt < 8; ++nt) acc[nt] = wmma_af<128>(A, WrS, 128, nt * 16, lane, acc[nt]);
        }
        {
            AF<128> A = load_af<128>(Asb, 128, lane);
#pragma unroll
            for (int nt = 0; nt < 8; ++nt) acc[nt] = wmma_af<128>(A, WbS, 128, nt * 16, lane, acc[nt]);
        }
#pragma unroll
        for (int nt = 0; nt < 8; ++nt) {
            int col = nt * 16 + (lane & 15);
            float bias = blS[col] + brS[col] + bbS[col];
#pragma unroll
            for (int j = 0; j < 8; ++j)
                op[j][col] = acc[nt][j] + bias + aLp[j][col] + aRp[j][col];
        }
    }
    // vector part, per component
#pragma unroll
    for (int i = 0; i < 3; ++i) {
        v8f acc[4];
#pragma unroll
        for (int nt = 0; nt < 4; ++nt) acc[nt] = v8f{};
        {
            AF<64> A = load_af<64>(Avl + i * 1024, 64, lane);
#pragma unroll
            for (int nt = 0; nt < 4; ++nt) acc[nt] = wmma_af<64>(A, WlV, 64, nt * 16, lane, acc[nt]);
        }
        {
            AF<64> A = load_af<64>(Avr + i * 1024, 64, lane);
#pragma unroll
            for (int nt = 0; nt < 4; ++nt) acc[nt] = wmma_af<64>(A, WrV, 64, nt * 16, lane, acc[nt]);
        }
        {
            AF<64> A = load_af<64>(Avb + i * 1024, 64, lane);
#pragma unroll
            for (int nt = 0; nt < 4; ++nt) acc[nt] = wmma_af<64>(A, WbV, 64, nt * 16, lane, acc[nt]);
        }
#pragma unroll
        for (int nt = 0; nt < 4; ++nt) {
            int colw = nt * 16 + (lane & 15);
            int oc = 128 + colw * 3 + i;
#pragma unroll
            for (int j = 0; j < 8; ++j)
                op[j][oc] = acc[nt][j] + aLp[j][oc] + aRp[j][oc];
        }
    }
}

// ---------------- host ----------------

extern "C" void kernel_launch(void* const* d_in, const int* in_sizes, int n_in,
                              void* d_out, int out_size, void* d_ws, size_t ws_size,
                              hipStream_t stream)
{
    (void)in_sizes; (void)n_in; (void)out_size; (void)ws_size;
    const float* h     = (const float*)d_in[0];
    const float* eIn   = (const float*)d_in[1];
    const float* rbf   = (const float*)d_in[2];
    const float* eattr = (const float*)d_in[3];
    const int*   left  = (const int*)d_in[4];
    const int*   right = left + E;

    size_t cur = 0;
    auto alloc = [&](size_t bytes) -> void* {
        void* p = (char*)d_ws + cur;
        cur += (bytes + 255) & ~(size_t)255;
        return p;
    };
    float* aggL = (float*)alloc((size_t)NNODES * 320 * 4);
    float* aggR = (float*)alloc((size_t)NNODES * 320 * 4);
    bf16t* hbS  = (bf16t*)alloc((size_t)E * 128 * 2);
    bf16t* wSf1 = (bf16t*)alloc(160 * 256 * 2);
    bf16t* wSf2 = (bf16t*)alloc(256 * 128 * 2);
    bf16t *wBS[2], *wBV[2], *wNS[2], *wNV[2], *wF1S[2], *wF1V[2], *wF2S[2], *wF2V[2], *wG1[2];
    for (int f = 0; f < 2; ++f) {
        wBS[f]  = (bf16t*)alloc(128 * 64 * 2);
        wBV[f]  = (bf16t*)alloc(64 * 32 * 2);
        wNS[f]  = (bf16t*)alloc(128 * 64 * 2);
        wNV[f]  = (bf16t*)alloc(64 * 32 * 2);
        wF1S[f] = (bf16t*)alloc(160 * 128 * 2);
        wF1V[f] = (bf16t*)alloc(96 * 64 * 2);
        wF2S[f] = (bf16t*)alloc(128 * 128 * 2);
        wF2V[f] = (bf16t*)alloc(64 * 64 * 2);
        wG1[f]  = (bf16t*)alloc(288 * 272 * 2);
    }
    bf16t* wNlS = (bf16t*)alloc(128 * 128 * 2);
    bf16t* wNlV = (bf16t*)alloc(64 * 64 * 2);
    bf16t* wNrS = (bf16t*)alloc(128 * 128 * 2);
    bf16t* wNrV = (bf16t*)alloc(64 * 64 * 2);
    bf16t* wSeS = (bf16t*)alloc(128 * 128 * 2);
    bf16t* wSeV = (bf16t*)alloc(64 * 64 * 2);

    auto cw = [&](int di, bf16t* dst, int K, int N, int kpad, int npad) {
        int total = kpad * npad;
        conv_w<<<(total + 255) / 256, 256, 0, stream>>>((const float*)d_in[di], dst, K, N, kpad, npad);
    };
    cw(5, wSf1, 160, 256, 160, 256);
    cw(7, wSf2, 256, 128, 256, 128);
    for (int f = 0; f < 2; ++f) {
        int b = 9 + 17 * f;
        cw(b + 0,  wBS[f],  128, 64, 128, 64);
        cw(b + 2,  wBV[f],  64, 32, 64, 32);
        cw(b + 3,  wNS[f],  128, 64, 128, 64);
        cw(b + 5,  wNV[f],  64, 32, 64, 32);
        cw(b + 6,  wF1S[f], 129, 128, 160, 128);
        cw(b + 8,  wF1V[f], 65, 64, 96, 64);
        cw(b + 9,  wF2S[f], 128, 128, 128, 128);
        cw(b + 11, wF2V[f], 64, 64, 64, 64);
        cw(b + 12, wG1[f],  257, 257, 288, 272);
    }
    cw(43, wNlS, 128, 128, 128, 128);
    cw(45, wNlV, 64, 64, 64, 64);
    cw(46, wNrS, 128, 128, 128, 128);
    cw(48, wNrV, 64, 64, 64, 64);
    cw(49, wSeS, 128, 128, 128, 128);
    cw(51, wSeV, 64, 64, 64, 64);

    // aggL and aggR are contiguous (both sizes are multiples of 256B)
    hipMemsetAsync(aggL, 0, (size_t)NNODES * 320 * 4 * 2, stream);

    fused_mlp_kernel<<<E / 64, 128, 0, stream>>>(eIn, rbf, wSf1, (const float*)d_in[6],
                                                 wSf2, (const float*)d_in[8], hbS);

    FfnW Wf[2];
    for (int f = 0; f < 2; ++f) {
        int b = 9 + 17 * f;
        Wf[f] = FfnW{ wBS[f], wBV[f], wNS[f], wNV[f], wF1S[f], wF1V[f], wF2S[f], wF2V[f], wG1[f],
                      (const float*)d_in[b + 1],  (const float*)d_in[b + 4],
                      (const float*)d_in[b + 7],  (const float*)d_in[b + 10],
                      (const float*)d_in[b + 13], (const float*)d_in[b + 15],
                      (const float*)d_in[b + 16] };
    }
    // ffn_l: gather at left, aggregate by right;  ffn_r: gather at right, aggregate by left
    bond_ffn_kernel<<<E / 16, 32, 0, stream>>>(hbS, eIn, h, eattr, left,  right, aggL, Wf[0]);
    bond_ffn_kernel<<<E / 16, 32, 0, stream>>>(hbS, eIn, h, eattr, right, left,  aggR, Wf[1]);

    finalize_kernel<<<E / 32, 64, 0, stream>>>(h, hbS, eIn, left, right, aggL, aggR,
        wNlS, (const float*)d_in[44], wNlV,
        wNrS, (const float*)d_in[47], wNrV,
        wSeS, (const float*)d_in[50], wSeV,
        (float*)d_out);
}
